// HeteroGraphSATEncoder_20796231647275
// MI455X (gfx1250) — matrix operand order repaired and checked by reading, not compiled, not run on hardware
//
#include <hip/hip_runtime.h>
#include <math.h>
#include <stdint.h>

typedef __attribute__((ext_vector_type(16))) _Float16 v16h;
typedef __attribute__((ext_vector_type(8)))  float    v8f;

#define HIDDEN 128
#define HEADS  4
#define HC     512   // HEADS * HIDDEN

__device__ __forceinline__ float gelu_exact(float x){
  return 0.5f * x * (1.0f + erff(x * 0.7071067811865475f));
}
// order-preserving float -> uint mapping for atomicMax-based segment max
__device__ __forceinline__ unsigned mono_f32(float x){
  unsigned u = __float_as_uint(x);
  return (u & 0x80000000u) ? ~u : (u | 0x80000000u);
}
__device__ __forceinline__ float demono_f32(unsigned u){
  return (u & 0x80000000u) ? __uint_as_float(u & 0x7fffffffu)
                           : __uint_as_float(~u);
}

// CDNA5 async global->LDS copy (ASYNCcnt-tracked), 16B per lane.
// LDS destination address is taken from the vdst VGPR; generic shared-pointer low 32 bits are the LDS offset.
__device__ __forceinline__ void async_load_b128(unsigned lds_off, const float* gptr){
  asm volatile("global_load_async_to_lds_b128 %0, %1, off"
               :: "v"(lds_off), "v"((unsigned long long)(uintptr_t)gptr)
               : "memory");
}
__device__ __forceinline__ void wait_async0(){
  asm volatile("s_wait_asynccnt 0x0" ::: "memory");
}

__global__ void fill_f32_kernel(float* __restrict__ p, float v, long n){
  long i = (long)blockIdx.x * blockDim.x + threadIdx.x;
  if (i < n) p[i] = v;
}

// C[M,N] (+)= A[M,K] @ B[K,N] (+ bias[N]); fp32 in/out, f16 WMMA compute.
// flags: bit0 = add bias, bit1 = accumulate into existing C.
// 256 threads = 8 wave32; block tile 32x128; each wave owns two 16x16 D tiles
// (shared A fragment). Tiles staged f32 in LDS via async global->LDS B128 DMA.
__global__ __launch_bounds__(256) void gemm_f32_wmma_kernel(
    const float* __restrict__ A, const float* __restrict__ B,
    const float* __restrict__ bias, float* __restrict__ C,
    int M, int N, int K, int flags)
{
  __shared__ float As[32][36];    // [m][k], row stride 144B (16B-aligned)
  __shared__ float Bs[32][132];   // [k][n], row stride 528B (16B-aligned)
  const int tid  = threadIdx.x;
  const int wave = tid >> 5;
  const int lane = tid & 31;
  const int half = lane >> 4;       // lane 0-15 vs 16-31
  const int l16  = lane & 15;
  const int tm   = wave >> 2;       // 0..1 -> M sub-tile
  const int tn   = wave & 3;        // 0..3 -> N sub-tile pair base
  const int blockM = blockIdx.y * 32;
  const int blockN = blockIdx.x * 128;

  if (K & 31) {                      // K-tail (K=16): zero-pad stays in LDS
    for (int t = tid; t < 32*36;  t += 256) ((float*)As)[t] = 0.0f;
    for (int t = tid; t < 32*132; t += 256) ((float*)Bs)[t] = 0.0f;
  }

  v8f acc0 = {}, acc1 = {};
  for (int k0 = 0; k0 < K; k0 += 32) {
    __syncthreads();
    // A tile: 32x32 f32 = 256 16B groups, one async B128 per thread
    {
      int r  = tid >> 3, kq = (tid & 7) * 4;
      int gr = blockM + r, gk = k0 + kq;
      if (gr < M && gk < K)
        async_load_b128((unsigned)(uintptr_t)&As[r][kq], A + (long)gr * K + gk);
    }
    // B tile: 32x128 f32 = 1024 16B groups, four async B128 per thread
#pragma unroll
    for (int j = 0; j < 4; j++) {
      int g  = tid + 256 * j;
      int kk = g >> 5, cq = (g & 31) * 4;
      int gk = k0 + kk, gc = blockN + cq;
      if (gk < K && gc < N)
        async_load_b128((unsigned)(uintptr_t)&Bs[kk][cq], B + (long)gk * N + gc);
    }
    wait_async0();       // drain this wave's async copies
    __syncthreads();     // all waves' tiles now visible

    // Build fragments in contiguous k order (k = 16*half + s), same for A and B (f32 -> f16 on the fly).
    v16h af, bf0, bf1;
#pragma unroll
    for (int i = 0; i < 8; i++) {
      // A 16x32 f16: lanes0-15 K=0..15, lanes16-31 K=16..31; vgpr i holds K=2i,2i+1
      int ka = half * 16 + 2 * i;
      af[2*i]   = (_Float16)As[tm*16 + l16][ka];
      af[2*i+1] = (_Float16)As[tm*16 + l16][ka + 1];
      // B 32x16 f16: lanes0-15 K=0..15, lanes16-31 K=16..31; vgpr i holds K=2i,2i+1
      int kb = half * 16 + 2 * i;
      bf0[2*i]   = (_Float16)Bs[kb][tn*16 + l16];
      bf0[2*i+1] = (_Float16)Bs[kb + 1][tn*16 + l16];
      bf1[2*i]   = (_Float16)Bs[kb][64 + tn*16 + l16];
      bf1[2*i+1] = (_Float16)Bs[kb + 1][64 + tn*16 + l16];
    }
    acc0 = __builtin_amdgcn_wmma_f32_16x16x32_f16(
        false, af, false, bf0, (short)0, acc0, false, false);
    acc1 = __builtin_amdgcn_wmma_f32_16x16x32_f16(
        false, af, false, bf1, (short)0, acc1, false, false);
  }

  // D 16x16 f32: vgpr r -> M = r + 8*half; N = lane&15
  const int col0 = blockN + tn*16 + l16;
  const int col1 = col0 + 64;
#pragma unroll
  for (int r = 0; r < 8; r++) {
    int row = blockM + tm*16 + half*8 + r;
    if (row < M) {
      if (col0 < N) {
        float v = acc0[r];
        if (flags & 1) v += bias[col0];
        long idx = (long)row * N + col0;
        if (flags & 2) v += C[idx];
        C[idx] = v;
      }
      if (col1 < N) {
        float v = acc1[r];
        if (flags & 1) v += bias[col1];
        long idx = (long)row * N + col1;
        if (flags & 2) v += C[idx];
        C[idx] = v;
      }
    }
  }
}

// ---- SAGE aggregation ----
__global__ void sage_count_kernel(const int* __restrict__ dst, float* __restrict__ cnt, int E){
  int e = blockIdx.x * blockDim.x + threadIdx.x;
  if (e < E) atomicAdd(&cnt[dst[e]], 1.0f);
}
__global__ void sage_scatter_kernel(const float* __restrict__ hsrc,
                                    const int* __restrict__ src, const int* __restrict__ dst,
                                    float* __restrict__ msum, int E){
  long i = (long)blockIdx.x * blockDim.x + threadIdx.x;
  if (i >= (long)E * HIDDEN) return;
  int e = (int)(i >> 7), c = (int)(i & 127);
  atomicAdd(&msum[(long)dst[e]*HIDDEN + c], hsrc[(long)src[e]*HIDDEN + c]);
}
__global__ void sage_mean_kernel(float* __restrict__ msum, const float* __restrict__ cnt, int Nd){
  long i = (long)blockIdx.x * blockDim.x + threadIdx.x;
  if (i >= (long)Nd * HIDDEN) return;
  msum[i] /= fmaxf(cnt[(int)(i >> 7)], 1.0f);
}

// ---- residual + LayerNorm + exact GELU: one wave32 per 128-wide row ----
__global__ void ln_gelu_kernel(float* __restrict__ h, const float* __restrict__ o,
                               const float* __restrict__ g, const float* __restrict__ b,
                               int Nrows){
  long gtid = (long)blockIdx.x * blockDim.x + threadIdx.x;
  int row  = (int)(gtid >> 5);
  int lane = threadIdx.x & 31;
  if (row >= Nrows) return;                 // wave-uniform exit
  long base = (long)row * HIDDEN;
  float v[4]; float sum = 0.0f;
#pragma unroll
  for (int i = 0; i < 4; i++){ int c = lane + 32*i; v[i] = h[base+c] + o[base+c]; sum += v[i]; }
#pragma unroll
  for (int m = 16; m > 0; m >>= 1) sum += __shfl_xor(sum, m, 32);
  float mu = sum * (1.0f/128.0f);
  float var = 0.0f;
#pragma unroll
  for (int i = 0; i < 4; i++){ float d = v[i] - mu; var += d*d; }
#pragma unroll
  for (int m = 16; m > 0; m >>= 1) var += __shfl_xor(var, m, 32);
  var *= (1.0f/128.0f);
  float inv = rsqrtf(var + 1e-5f);
#pragma unroll
  for (int i = 0; i < 4; i++){
    int c = lane + 32*i;
    float y = (v[i] - mu) * inv * g[c] + b[c];
    h[base+c] = gelu_exact(y);
  }
}

// ---- GATv2 ----
__global__ void gat_alpha_kernel(const float* __restrict__ xl, const float* __restrict__ xr,
                                 const float* __restrict__ ea, const float* __restrict__ We,
                                 const float* __restrict__ att,
                                 const int* __restrict__ src, const int* __restrict__ dst,
                                 float* __restrict__ alpha, unsigned* __restrict__ mbuf,
                                 int E, int edim){
  int i = blockIdx.x * blockDim.x + threadIdx.x;
  if (i >= E * HEADS) return;
  int e = i >> 2, hh = i & 3;
  int s = src[e], d = dst[e];
  const float* xls = xl + (long)s * HC + hh * HIDDEN;
  const float* xrd = xr + (long)d * HC + hh * HIDDEN;
  float eav[5];
  for (int k = 0; k < edim; k++) eav[k] = ea[(long)e * edim + k];
  float a = 0.0f;
  for (int c = 0; c < HIDDEN; c++){
    float ep = 0.0f;
    for (int k = 0; k < edim; k++) ep += eav[k] * We[(long)k * HC + hh * HIDDEN + c];
    float gv = xls[c] + xrd[c] + ep;
    gv = (gv > 0.0f) ? gv : 0.2f * gv;          // leaky_relu 0.2
    a += att[hh * HIDDEN + c] * gv;
  }
  alpha[i] = a;
  atomicMax(&mbuf[d * HEADS + hh], mono_f32(a));
}
__global__ void gat_exp_kernel(float* __restrict__ alpha, const unsigned* __restrict__ mbuf,
                               float* __restrict__ sbuf, const int* __restrict__ dst, int E){
  int i = blockIdx.x * blockDim.x + threadIdx.x;
  if (i >= E * HEADS) return;
  int e = i >> 2, hh = i & 3;
  int d = dst[e];
  float a = expf(alpha[i] - demono_f32(mbuf[d * HEADS + hh]));
  alpha[i] = a;
  atomicAdd(&sbuf[d * HEADS + hh], a);
}
__global__ void gat_agg_kernel(const float* __restrict__ xl, const float* __restrict__ alpha,
                               const float* __restrict__ sbuf,
                               const int* __restrict__ src, const int* __restrict__ dst,
                               float* __restrict__ agg, int E){
  long i = (long)blockIdx.x * blockDim.x + threadIdx.x;
  if (i >= (long)E * HC) return;
  int e = (int)(i >> 9), c = (int)(i & 511);
  int hh = c >> 7;
  int s = src[e], d = dst[e];
  float w = alpha[e * HEADS + hh] / (sbuf[d * HEADS + hh] + 1e-16f);
  atomicAdd(&agg[(long)d * HC + c], w * xl[(long)s * HC + c]);
}
__global__ void gat_headmean_kernel(const float* __restrict__ agg, const float* __restrict__ bias,
                                    float* __restrict__ o, int Nd){
  long i = (long)blockIdx.x * blockDim.x + threadIdx.x;
  if (i >= (long)Nd * HIDDEN) return;
  int n = (int)(i >> 7), c = (int)(i & 127);
  long b = (long)n * HC + c;
  float v = 0.25f * (agg[b] + agg[b + 128] + agg[b + 256] + agg[b + 384]);
  o[i] += v + bias[c];
}

extern "C" void kernel_launch(void* const* d_in, const int* in_sizes, int n_in,
                              void* d_out, int out_size, void* d_ws, size_t ws_size,
                              hipStream_t stream)
{
  (void)in_sizes; (void)n_in; (void)out_size; (void)ws_size;

  const int NN[3]   = {30000, 15000, 6000};        // flow, time, attr
  const int IND[3]  = {64, 32, 16};                // input dims
  const int eiIdx[4]= {3, 5, 7, 9};
  const int eaIdx[4]= {4, 6, 8, 10};
  const int eSrc[4] = {0, 1, 0, 0};                // acts_in, evolves_to, interacts, uses
  const int eDst[4] = {1, 1, 0, 2};
  const int eDim[4] = {2, 2, 5, 1};
  const int eNum[4] = {100000, 60000, 100000, 60000};
  const int ntSort[3] = {1, 2, 0};                 // flow/time/attr -> sorted (attr,flow,time) slot

  // params flattened (jax tree order: sorted dict keys; 'W*' < lowercase):
  // gat@0: l*28+et*7+{We,Wl,Wr,att,bias,bl,br}; gat_norm@56: l*6+nt*2+{b,g};
  // in@68: nt*2+{W,b}; out@74; sage@80: l*12+et*3+{Wl,Wr,bl}; sage_norm@104: l*6+nt*2+{b,g}
  const float* P[116];
  for (int i = 0; i < 116; i++) P[i] = (const float*)d_in[11 + i];

  // workspace carve-out
  char* wp = (char*)d_ws;
  size_t off = 0;
  auto alloc = [&](size_t bytes) -> char* {
    char* p = wp + off;
    off = (off + bytes + 255) & ~(size_t)255;
    return p;
  };
  float* h[3]; float* o[3];
  for (int t = 0; t < 3; t++) h[t] = (float*)alloc((size_t)NN[t]*HIDDEN*4);
  for (int t = 0; t < 3; t++) o[t] = (float*)alloc((size_t)NN[t]*HIDDEN*4);
  float*    msum = (float*)alloc((size_t)30000*HIDDEN*4);
  float*    cnt  = (float*)alloc((size_t)30000*4);
  float*    xl   = (float*)alloc((size_t)30000*HC*4);
  float*    xr   = (float*)alloc((size_t)30000*HC*4);
  float*    agg  = (float*)alloc((size_t)30000*HC*4);
  float*    alp  = (float*)alloc((size_t)100000*HEADS*4);
  unsigned* mb   = (unsigned*)alloc((size_t)30000*HEADS*4);
  float*    sb   = (float*)alloc((size_t)30000*HEADS*4);

  auto fill = [&](float* p, float v, long n){
    fill_f32_kernel<<<(int)((n + 255) / 256), 256, 0, stream>>>(p, v, n);
  };
  auto gemm = [&](const float* A, const float* B, const float* bias, float* C,
                  int M, int N, int K, int flags){
    dim3 g((N + 127) / 128, (M + 31) / 32);
    gemm_f32_wmma_kernel<<<g, 256, 0, stream>>>(A, B, bias, C, M, N, K, flags);
  };

  // input projections
  for (int t = 0; t < 3; t++)
    gemm((const float*)d_in[t], P[68 + ntSort[t]*2], P[68 + ntSort[t]*2 + 1],
         h[t], NN[t], HIDDEN, IND[t], 1);

  // SAGE layers
  for (int l = 0; l < 2; l++){
    for (int t = 0; t < 3; t++) fill(o[t], 0.0f, (long)NN[t]*HIDDEN);
    for (int et = 0; et < 4; et++){
      int s = eSrc[et], d = eDst[et], E = eNum[et];
      const int* src = (const int*)d_in[eiIdx[et]];
      const int* dst = src + E;
      fill(msum, 0.0f, (long)NN[d]*HIDDEN);
      fill(cnt,  0.0f, NN[d]);
      sage_count_kernel<<<(E + 255)/256, 256, 0, stream>>>(dst, cnt, E);
      long ns = (long)E * HIDDEN;
      sage_scatter_kernel<<<(int)((ns + 255)/256), 256, 0, stream>>>(h[s], src, dst, msum, E);
      long nm = (long)NN[d] * HIDDEN;
      sage_mean_kernel<<<(int)((nm + 255)/256), 256, 0, stream>>>(msum, cnt, NN[d]);
      int pb = 80 + l*12 + et*3;
      gemm(msum, P[pb+0], P[pb+2], o[d], NN[d], HIDDEN, HIDDEN, 1 | 2);  // mean@Wl + bl
      gemm(h[d], P[pb+1], nullptr, o[d], NN[d], HIDDEN, HIDDEN, 2);      // + x@Wr
    }
    for (int t = 0; t < 3; t++){
      int pb = 104 + l*6 + ntSort[t]*2;
      ln_gelu_kernel<<<(NN[t] + 7)/8, 256, 0, stream>>>(h[t], o[t], P[pb+1], P[pb+0], NN[t]);
    }
  }

  // GATv2 layers
  for (int l = 0; l < 2; l++){
    for (int t = 0; t < 3; t++) fill(o[t], 0.0f, (long)NN[t]*HIDDEN);
    for (int et = 0; et < 4; et++){
      int s = eSrc[et], d = eDst[et], E = eNum[et], ed = eDim[et];
      const int* src = (const int*)d_in[eiIdx[et]];
      const int* dst = src + E;
      const float* ea = (const float*)d_in[eaIdx[et]];
      int pb = l*28 + et*7;
      const float *We = P[pb+0], *Wl = P[pb+1], *Wr = P[pb+2], *att = P[pb+3];
      const float *bia = P[pb+4], *bl = P[pb+5], *br = P[pb+6];
      gemm(h[s], Wl, bl, xl, NN[s], HC, HIDDEN, 1);
      gemm(h[d], Wr, br, xr, NN[d], HC, HIDDEN, 1);
      fill((float*)mb, 0.0f, (long)NN[d]*HEADS);       // bits 0 sort below mono(-inf)
      fill(sb,         0.0f, (long)NN[d]*HEADS);
      fill(agg,        0.0f, (long)NN[d]*HC);
      int na = E * HEADS;
      gat_alpha_kernel<<<(na + 255)/256, 256, 0, stream>>>(xl, xr, ea, We, att, src, dst, alp, mb, E, ed);
      gat_exp_kernel<<<(na + 255)/256, 256, 0, stream>>>(alp, mb, sb, dst, E);
      long ng = (long)E * HC;
      gat_agg_kernel<<<(int)((ng + 255)/256), 256, 0, stream>>>(xl, alp, sb, src, dst, agg, E);
      long nh = (long)NN[d] * HIDDEN;
      gat_headmean_kernel<<<(int)((nh + 255)/256), 256, 0, stream>>>(agg, bia, o[d], NN[d]);
    }
    for (int t = 0; t < 3; t++){
      int pb = 56 + l*6 + ntSort[t]*2;
      ln_gelu_kernel<<<(NN[t] + 7)/8, 256, 0, stream>>>(h[t], o[t], P[pb+1], P[pb+0], NN[t]);
    }
  }

  // output projections straight into d_out (flow | time | attr)
  float* outp = (float*)d_out;
  long outOff[3] = {0, (long)30000*HIDDEN, (long)(30000 + 15000)*HIDDEN};
  for (int t = 0; t < 3; t++)
    gemm(h[t], P[74 + ntSort[t]*2], P[74 + ntSort[t]*2 + 1],
         outp + outOff[t], NN[t], HIDDEN, HIDDEN, 1);
}
